// MergeNN_81862076662054
// MI455X (gfx1250) — matrix operand order repaired and checked by reading, not compiled, not run on hardware
//
#include <hip/hip_runtime.h>
#include <hip/hip_bf16.h>

typedef __attribute__((ext_vector_type(16))) __bf16 v16bf;
typedef __attribute__((ext_vector_type(8)))  float  v8f;

#define NN     60000
#define KK     1024
#define DD     784
#define DPAD   800      // 25 * 32
#define DMAIN  768      // 24 full 32-wide K steps; tail step covers 768..783
#define CC     10
#define NCHUNK 25       // n-split for k_match / k_rbf grids
#define IDXMASK 0x1FFFFu      // 17 bits: NN < 131072
#define KEYMASK 0xFFFE0000u   // top 15 float bits (monotone for d2 >= 0)

// ---------------- fragment builders (contiguous k = 16*half + i order, wave32) ----------------

__device__ inline v16bf frag_a_lds(const __bf16* __restrict__ s, int ld, int lane, int ds) {
  int m = lane & 15, half = lane >> 4;
  const __bf16* r = s + m * ld + ds;
  v16bf a;
#pragma unroll
  for (int v = 0; v < 16; ++v) a[v] = r[half*16 + v];
  return a;
}

// B[k][n] = F[n0+n][ds+k], fp32 source, no bounds checks (valid for ds <= 752).
__device__ inline v16bf frag_b_full(const float* __restrict__ F, int n0, int ds, int lane) {
  int nn = lane & 15, half = lane >> 4;
  const float* p = F + (long long)(n0 + nn) * DD + ds + half * 16;
  v16bf b;
#pragma unroll
  for (int i = 0; i < 16; ++i) b[i] = (__bf16)p[i];
  return b;
}

// Tail K-step (768..799): half==0 lanes carry columns 768..783, half==1 lanes zero.
__device__ inline v16bf frag_b_tail(const float* __restrict__ F, int n0, int lane) {
  int nn = lane & 15, half = lane >> 4;
  v16bf b;
  if (half == 0) {
    const float* p = F + (long long)(n0 + nn) * DD + DMAIN;
#pragma unroll
    for (int i = 0; i < 16; ++i) b[i] = (__bf16)p[i];
  } else {
#pragma unroll
    for (int i = 0; i < 16; ++i) b[i] = (__bf16)0.0f;
  }
  return b;
}

#define WMMA_BF16(A, B, C) \
  __builtin_amdgcn_wmma_f32_16x16x32_bf16(false, (A), false, (B), (short)0, (C), false, false)

__device__ inline unsigned packkey(float d2, int idx) {
  float c = fmaxf(d2, 0.0f);
  return (__float_as_uint(c) & KEYMASK) | (unsigned)idx;
}

// ---------------- kernels ----------------

__global__ void k_rownorm(const float* __restrict__ A, float* __restrict__ out, int rows) {
  int wave = threadIdx.x >> 5, lane = threadIdx.x & 31;
  int row = blockIdx.x * 8 + wave;
  if (row >= rows) return;
  const float* p = A + (long long)row * DD;
  float s = 0.f;
  for (int j = lane; j < DD; j += 32) { float v = p[j]; s += v * v; }
#pragma unroll
  for (int m = 16; m; m >>= 1) s += __shfl_xor(s, m, 32);
  if (lane == 0) out[row] = s;
}

// Partial argmin_n ||x_k - F_n||^2 with packed (d2,idx) keys; min-reduce is one v_min_u32.
__global__ void __launch_bounds__(256, 1)
k_match(const float* __restrict__ X, const float* __restrict__ F,
        const float* __restrict__ xn, const float* __restrict__ fn,
        unsigned* __restrict__ pkey) {
  __shared__ __bf16 sA[2][16 * DPAD];
  __shared__ unsigned rk[8][32];
  int t = threadIdx.x, lane = t & 31, wave = t >> 5;
  int kt0 = blockIdx.x * 2;          // rows kt0*16 .. kt0*16+31

  for (int i = t; i < 32 * DPAD; i += blockDim.x) {
    int m = i / DPAD, j = i % DPAD;
    float v = (j < DD) ? X[(long long)(kt0 * 16 + m) * DD + j] : 0.0f;
    sA[m >> 4][(m & 15) * DPAD + j] = (__bf16)v;
  }
  __syncthreads();

  int nl = lane & 15, half = lane >> 4, mbase = half * 8;
  float xnv0[8], xnv1[8];
#pragma unroll
  for (int v = 0; v < 8; ++v) {
    xnv0[v] = xn[kt0 * 16 + mbase + v];
    xnv1[v] = xn[kt0 * 16 + 16 + mbase + v];
  }
  unsigned key0[8], key1[8];
#pragma unroll
  for (int v = 0; v < 8; ++v) { key0[v] = 0xFFFFFFFFu; key1[v] = 0xFFFFFFFFu; }

  int pairs = NN / 32;
  int per = (pairs + gridDim.y - 1) / gridDim.y;
  int p0 = blockIdx.y * per, p1 = p0 + per; if (p1 > pairs) p1 = pairs;

  for (int p = p0 + wave; p < p1; p += 8) {
    int n0 = p * 32;
    if (p + 8 < p1) {
      __builtin_prefetch(F + (long long)((p + 8) * 32 + nl) * DD, 0, 3);
      __builtin_prefetch(F + (long long)((p + 8) * 32 + 16 + nl) * DD, 0, 3);
    }
    v8f s00 = {}, s01 = {}, s10 = {}, s11 = {};
#pragma unroll 2
    for (int ds = 0; ds < DMAIN; ds += 32) {
      v16bf b0 = frag_b_full(F, n0,      ds, lane);
      v16bf b1 = frag_b_full(F, n0 + 16, ds, lane);
      v16bf a0 = frag_a_lds(sA[0], DPAD, lane, ds);
      v16bf a1 = frag_a_lds(sA[1], DPAD, lane, ds);
      s00 = WMMA_BF16(a0, b0, s00); s01 = WMMA_BF16(a0, b1, s01);
      s10 = WMMA_BF16(a1, b0, s10); s11 = WMMA_BF16(a1, b1, s11);
    }
    { // tail K-step
      v16bf b0 = frag_b_tail(F, n0, lane);
      v16bf b1 = frag_b_tail(F, n0 + 16, lane);
      v16bf a0 = frag_a_lds(sA[0], DPAD, lane, DMAIN);
      v16bf a1 = frag_a_lds(sA[1], DPAD, lane, DMAIN);
      s00 = WMMA_BF16(a0, b0, s00); s01 = WMMA_BF16(a0, b1, s01);
      s10 = WMMA_BF16(a1, b0, s10); s11 = WMMA_BF16(a1, b1, s11);
    }
    float fn0 = fn[n0 + nl], fn1 = fn[n0 + 16 + nl];
    int i0 = n0 + nl, i1 = n0 + 16 + nl;
#pragma unroll
    for (int v = 0; v < 8; ++v) {
      key0[v] = min(key0[v], packkey(xnv0[v] + fn0 - 2.0f * s00[v], i0));
      key0[v] = min(key0[v], packkey(xnv0[v] + fn1 - 2.0f * s01[v], i1));
      key1[v] = min(key1[v], packkey(xnv1[v] + fn0 - 2.0f * s10[v], i0));
      key1[v] = min(key1[v], packkey(xnv1[v] + fn1 - 2.0f * s11[v], i1));
    }
  }
#pragma unroll
  for (int off = 1; off < 16; off <<= 1) {
#pragma unroll
    for (int v = 0; v < 8; ++v) {
      key0[v] = min(key0[v], __shfl_xor(key0[v], off, 32));
      key1[v] = min(key1[v], __shfl_xor(key1[v], off, 32));
    }
  }
  if (nl == 0) {
#pragma unroll
    for (int v = 0; v < 8; ++v) {
      rk[wave][mbase + v]      = key0[v];
      rk[wave][16 + mbase + v] = key1[v];
    }
  }
  __syncthreads();
  if (t < 32) {
    unsigned bk = rk[0][t];
#pragma unroll
    for (int w = 1; w < 8; ++w) bk = min(bk, rk[w][t]);
    pkey[blockIdx.y * KK + kt0 * 16 + t] = bk;
  }
}

__global__ void k_matchred(const unsigned* __restrict__ pkey, int* __restrict__ midx) {
  int r = blockIdx.x * blockDim.x + threadIdx.x;
  if (r >= KK) return;
  unsigned bk = 0xFFFFFFFFu;
#pragma unroll
  for (int c = 0; c < NCHUNK; ++c) bk = min(bk, pkey[c * KK + r]);
  midx[r] = (int)(bk & IDXMASK);
}

// gather xt=feats[midx], stage bf16 (padded), row norms, linear layer + nearest one-hot class
__global__ void k_prep(const int* __restrict__ midx, const float* __restrict__ feats,
                       const float* __restrict__ W, const float* __restrict__ bias,
                       const float* __restrict__ uls,
                       __bf16* __restrict__ xtb, float* __restrict__ xtn, int* __restrict__ cls) {
  int wave = threadIdx.x >> 5, lane = threadIdx.x & 31;
  int k = blockIdx.x * 8 + wave;
  int mi = midx[k];
  const float* p = feats + (long long)mi * DD;
  float y[CC];
#pragma unroll
  for (int c = 0; c < CC; ++c) y[c] = 0.f;
  float nrm = 0.f;
  for (int j = lane; j < DPAD; j += 32) {
    float v = (j < DD) ? p[j] : 0.0f;
    xtb[(long long)k * DPAD + j] = (__bf16)v;
    nrm += v * v;
    if (j < DD) {
      const float* wr = W + (long long)j * CC;
#pragma unroll
      for (int c = 0; c < CC; ++c) y[c] += v * wr[c];
    }
  }
#pragma unroll
  for (int m = 16; m; m >>= 1) {
    nrm += __shfl_xor(nrm, m, 32);
#pragma unroll
    for (int c = 0; c < CC; ++c) y[c] += __shfl_xor(y[c], m, 32);
  }
  if (lane == 0) {
    xtn[k] = nrm;
#pragma unroll
    for (int c = 0; c < CC; ++c) y[c] += bias[c];
    float best = 3.4e38f; int bc = 0;
#pragma unroll
    for (int c = 0; c < CC; ++c) {
      float s = 0.f;
#pragma unroll
      for (int j = 0; j < CC; ++j) { float t2 = y[j] - uls[c * CC + j]; s += t2 * t2; }
      if (s < best) { best = s; bc = c; }
    }
    cls[k] = bc;
  }
}

// RBF accumulation: two k-tiles per block, n split over blockIdx.y.
__global__ void __launch_bounds__(256, 1)
k_rbf(const __bf16* __restrict__ xtb, const float* __restrict__ xtn,
      const int* __restrict__ cls, const float* __restrict__ feats,
      const float* __restrict__ fn, const int* __restrict__ lab,
      const float* __restrict__ Ys, float* __restrict__ num) {
  __shared__ __bf16 sA[2][16 * DPAD];
  __shared__ __bf16 sW[8][16 * 32];
  int t = threadIdx.x, lane = t & 31, wave = t >> 5;
  int kt0 = blockIdx.x * 2;

  for (int i = t; i < 32 * DPAD; i += blockDim.x)
    ((__bf16*)sA)[i] = xtb[(long long)kt0 * 16 * DPAD + i];
  __syncthreads();

  int nl = lane & 15, half = lane >> 4, mbase = half * 8;
  float xnv0[8], xnv1[8]; int clv0[8], clv1[8];
#pragma unroll
  for (int v = 0; v < 8; ++v) {
    xnv0[v] = xtn[kt0 * 16 + mbase + v];      clv0[v] = cls[kt0 * 16 + mbase + v];
    xnv1[v] = xtn[kt0 * 16 + 16 + mbase + v]; clv1[v] = cls[kt0 * 16 + 16 + mbase + v];
  }

  v8f acc0 = {}, acc1 = {};
  int pairs = NN / 32;
  int per = (pairs + gridDim.y - 1) / gridDim.y;
  int p0 = blockIdx.y * per, p1 = p0 + per; if (p1 > pairs) p1 = pairs;
  __bf16* w = sW[wave];

  for (int p = p0 + wave; p < p1; p += 8) {
    int n0 = p * 32;
    if (p + 8 < p1) {
      __builtin_prefetch(feats + (long long)((p + 8) * 32 + nl) * DD, 0, 3);
      __builtin_prefetch(feats + (long long)((p + 8) * 32 + 16 + nl) * DD, 0, 3);
    }
    v8f s00 = {}, s01 = {}, s10 = {}, s11 = {};
#pragma unroll 2
    for (int ds = 0; ds < DMAIN; ds += 32) {
      v16bf b0 = frag_b_full(feats, n0,      ds, lane);
      v16bf b1 = frag_b_full(feats, n0 + 16, ds, lane);
      v16bf a0 = frag_a_lds(sA[0], DPAD, lane, ds);
      v16bf a1 = frag_a_lds(sA[1], DPAD, lane, ds);
      s00 = WMMA_BF16(a0, b0, s00); s01 = WMMA_BF16(a0, b1, s01);
      s10 = WMMA_BF16(a1, b0, s10); s11 = WMMA_BF16(a1, b1, s11);
    }
    { // tail K-step
      v16bf b0 = frag_b_tail(feats, n0, lane);
      v16bf b1 = frag_b_tail(feats, n0 + 16, lane);
      v16bf a0 = frag_a_lds(sA[0], DPAD, lane, DMAIN);
      v16bf a1 = frag_a_lds(sA[1], DPAD, lane, DMAIN);
      s00 = WMMA_BF16(a0, b0, s00); s01 = WMMA_BF16(a0, b1, s01);
      s10 = WMMA_BF16(a1, b0, s10); s11 = WMMA_BF16(a1, b1, s11);
    }
    float fn0 = fn[n0 + nl], fn1 = fn[n0 + 16 + nl];
    int   l0  = lab[n0 + nl], l1 = lab[n0 + 16 + nl];

    // B fragment for w @ [Y_star | 1 | 0...] -- shared by both k-tiles
    v16bf by;
    {
      int c = nl;
#pragma unroll
      for (int i = 0; i < 16; ++i) {
        int nn = n0 + half * 16 + i;
        float yv = (c < CC) ? Ys[(long long)nn * CC + c] : ((c == CC) ? 1.0f : 0.0f);
        by[i] = (__bf16)yv;
      }
    }

    // ---- epilogue, k-tile 0 ----
#pragma unroll
    for (int v = 0; v < 8; ++v) {
      float d20 = xnv0[v] + fn0 - 2.0f * s00[v];
      float d21 = xnv0[v] + fn1 - 2.0f * s01[v];
      float w0 = (l0 == clv0[v]) ? __expf(-d20) : 0.0f;
      float w1 = (l1 == clv0[v]) ? __expf(-d21) : 0.0f;
      w[(mbase + v) * 32 + nl]      = (__bf16)w0;
      w[(mbase + v) * 32 + 16 + nl] = (__bf16)w1;
    }
    __builtin_amdgcn_wave_barrier();   // wave-local LDS ordering (DS ops in-order per wave)
    {
      v16bf aw = frag_a_lds(w, 32, lane, 0);
      acc0 = WMMA_BF16(aw, by, acc0);
    }
    __builtin_amdgcn_wave_barrier();

    // ---- epilogue, k-tile 1 (reuse same per-wave LDS scratch) ----
#pragma unroll
    for (int v = 0; v < 8; ++v) {
      float d20 = xnv1[v] + fn0 - 2.0f * s10[v];
      float d21 = xnv1[v] + fn1 - 2.0f * s11[v];
      float w0 = (l0 == clv1[v]) ? __expf(-d20) : 0.0f;
      float w1 = (l1 == clv1[v]) ? __expf(-d21) : 0.0f;
      w[(mbase + v) * 32 + nl]      = (__bf16)w0;
      w[(mbase + v) * 32 + 16 + nl] = (__bf16)w1;
    }
    __builtin_amdgcn_wave_barrier();
    {
      v16bf aw = frag_a_lds(w, 32, lane, 0);
      acc1 = WMMA_BF16(aw, by, acc1);
    }
    __builtin_amdgcn_wave_barrier();
  }
#pragma unroll
  for (int v = 0; v < 8; ++v) {
    atomicAdd(&num[(kt0 * 16 + mbase + v) * 16 + nl],      acc0[v]);
    atomicAdd(&num[(kt0 * 16 + 16 + mbase + v) * 16 + nl], acc1[v]);
  }
}

__global__ void k_final(const float* __restrict__ n1, const float* __restrict__ n2,
                        float* __restrict__ out) {
  int i = blockIdx.x * blockDim.x + threadIdx.x;
  if (i >= KK * CC) return;
  int k = i / CC, c = i % CC;
  float a = n1[k * 16 + c] / n1[k * 16 + CC];
  float b = n2[k * 16 + c] / n2[k * 16 + CC];
  out[i] = 0.5f * (a + b);
}

// ---------------- launch ----------------

extern "C" void kernel_launch(void* const* d_in, const int* in_sizes, int n_in,
                              void* d_out, int out_size, void* d_ws, size_t ws_size,
                              hipStream_t stream) {
  const float* x      = (const float*)d_in[0];
  const float* Fstar  = (const float*)d_in[1];
  const float* Ystar  = (const float*)d_in[2];
  const float* feats1 = (const float*)d_in[3];
  const float* feats2 = (const float*)d_in[4];
  const float* uls1   = (const float*)d_in[5];
  const float* uls2   = (const float*)d_in[6];
  const float* W1     = (const float*)d_in[7];
  const float* b1     = (const float*)d_in[8];
  const float* W2     = (const float*)d_in[9];
  const float* b2     = (const float*)d_in[10];
  const int*   lab1   = (const int*)d_in[11];
  const int*   lab2   = (const int*)d_in[12];

  char* ws = (char*)d_ws;
  size_t off = 0;
  auto alloc = [&](size_t bytes) { char* p = ws + off; off += (bytes + 255) & ~size_t(255); return p; };
  float* fnS  = (float*)alloc(NN * 4);
  float* fn1  = (float*)alloc(NN * 4);
  float* fn2  = (float*)alloc(NN * 4);
  float* xn   = (float*)alloc(KK * 4);
  float* xtn1 = (float*)alloc(KK * 4);
  float* xtn2 = (float*)alloc(KK * 4);
  float* num1 = (float*)alloc(KK * 16 * 4);     // adjacent to num2 (both 256B multiples)
  float* num2 = (float*)alloc(KK * 16 * 4);
  unsigned* pkey = (unsigned*)alloc((size_t)NCHUNK * KK * 4);
  int*   midx = (int*)alloc(KK * 4);
  int*   cls1 = (int*)alloc(KK * 4);
  int*   cls2 = (int*)alloc(KK * 4);
  __bf16* xtb1 = (__bf16*)alloc((size_t)KK * DPAD * 2);
  __bf16* xtb2 = (__bf16*)alloc((size_t)KK * DPAD * 2);

  // row norms
  k_rownorm<<<(NN + 7) / 8, 256, 0, stream>>>(Fstar,  fnS, NN);
  k_rownorm<<<(NN + 7) / 8, 256, 0, stream>>>(feats1, fn1, NN);
  k_rownorm<<<(NN + 7) / 8, 256, 0, stream>>>(feats2, fn2, NN);
  k_rownorm<<<(KK + 7) / 8, 256, 0, stream>>>(x,      xn,  KK);

  // exact-match argmin via WMMA distance tiles (packed-key partials, then reduce)
  dim3 gm(KK / 32, NCHUNK);
  k_match<<<gm, 256, 0, stream>>>(x, Fstar, xn, fnS, pkey);
  k_matchred<<<(KK + 255) / 256, 256, 0, stream>>>(pkey, midx);

  // per-branch gather + linear + classify + bf16 staging
  k_prep<<<KK / 8, 256, 0, stream>>>(midx, feats1, W1, b1, uls1, xtb1, xtn1, cls1);
  k_prep<<<KK / 8, 256, 0, stream>>>(midx, feats2, W2, b2, uls2, xtb2, xtn2, cls2);

  hipMemsetAsync(num1, 0, (size_t)2 * KK * 16 * 4, stream);

  // RBF accumulation (32 k-tile-pair blocks x 25 n-chunks)
  dim3 gr(KK / 32, NCHUNK);
  k_rbf<<<gr, 256, 0, stream>>>(xtb1, xtn1, cls1, feats1, fn1, lab1, Ystar, num1);
  k_rbf<<<gr, 256, 0, stream>>>(xtb2, xtn2, cls2, feats2, fn2, lab2, Ystar, num2);

  k_final<<<(KK * CC + 255) / 256, 256, 0, stream>>>(num1, num2, (float*)d_out);
}
